// SimpleLSTM_10419590660203
// MI455X (gfx1250) — compile-verified, matrix operand order checked by reading, not run on hardware
//
#include <hip/hip_runtime.h>

// LSTM: B=128, T=512, D=256, H=512. Persistent kernel, weights resident in LDS.
// 64 blocks x 128 threads (4 waves) -> 1 wave per SIMD across 64 WGPs.
#define LB 128
#define LT 512
#define LD 256
#define LH 512
#define LK 768          // D + H
#define NBLK 64         // 32 column-slices x 2 row-halves
#define NTHR 128        // 4 waves
#define LDS_BYTES (24 * 4 * 32 * 16 * 2)   // [kk24][gate4][lane32][elem16] bf16 = 96 KB

typedef __attribute__((ext_vector_type(16))) __bf16 v16bf;
typedef __attribute__((ext_vector_type(8)))  __bf16 v8bf;
typedef __attribute__((ext_vector_type(8)))  float  v8f;

__device__ __forceinline__ float sigm(float x) { return 1.0f / (1.0f + __expf(-x)); }

// Branch-free tanh: sign(x) * (1 - e) / (1 + e), e = exp(-2|x|).
// One v_exp_f32 + a few VALU; exact at +-inf, monotone, ~1ulp-of-bf16 accurate.
__device__ __forceinline__ float ftanh(float x) {
  const float e = __expf(-2.0f * __builtin_fabsf(x));
  const float t = (1.0f - e) / (1.0f + e);
  return __builtin_copysignf(t, x);
}

__global__ void lstm_init(unsigned* cnt) { if (threadIdx.x == 0) *cnt = 0u; }

__global__ void __launch_bounds__(NTHR)
lstm_persist(const float* __restrict__ x, const float* __restrict__ W_ih,
             const float* __restrict__ W_hh, const float* __restrict__ b_ih,
             const float* __restrict__ b_hh, __bf16* __restrict__ hbuf,
             float* __restrict__ out, unsigned* __restrict__ cnt) {
  extern __shared__ __align__(32) __bf16 ldsW[];   // [24][4][32][16]

  const int tid   = threadIdx.x;
  const int lane  = tid & 31;
  const int wave  = tid >> 5;                 // 0..3
  const int j0    = (blockIdx.x >> 1) * 16;   // H column slice
  const int rbase = (blockIdx.x & 1) * 64;    // batch row half

  // ---- One-time: convert this block's weight slice fp32->bf16 into LDS,
  //      pre-swizzled into the WMMA B-operand lane layout.
  for (int idx = tid; idx < 24 * 4 * 32; idx += NTHR) {
    const int kk = idx >> 7;               // /128
    const int g  = (idx >> 5) & 3;
    const int ln = idx & 31;
    const int grow = g * LH + j0 + (ln & 15);
    const int ka = (ln < 16) ? 0 : 8;
    const int k0 = kk * 32 + ka;
    __bf16* dst = &ldsW[(size_t)idx * 16];
    const float* src = (k0 < LD) ? (W_ih + (size_t)grow * LD + k0)
                                 : (W_hh + (size_t)grow * LH + (k0 - LD));
    #pragma unroll
    for (int e = 0; e < 8; ++e) dst[e]     = (__bf16)src[e];
    #pragma unroll
    for (int e = 0; e < 8; ++e) dst[e + 8] = (__bf16)src[e + 16];
  }
  __syncthreads();

  // Fixed per-lane geometry (constant across all timesteps).
  const int rowA    = rbase + wave * 16 + (lane & 15);   // A-matrix row (batch)
  const int kaA     = (lane < 16) ? 0 : 8;               // A-matrix K sub-offset
  const int col     = j0 + (lane & 15);                  // D-matrix column (H index)
  const int rowbase = rbase + wave * 16 + ((lane < 16) ? 0 : 8);

  const float bi  = b_ih[0 * LH + col] + b_hh[0 * LH + col];
  const float bff = b_ih[1 * LH + col] + b_hh[1 * LH + col];
  const float bg  = b_ih[2 * LH + col] + b_hh[2 * LH + col];
  const float bo  = b_ih[3 * LH + col] + b_hh[3 * LH + col];

  float creg[8] = {0, 0, 0, 0, 0, 0, 0, 0};              // c-state stays in VGPRs

  const float* px0 = x + (size_t)rowA * LT * LD + kaA;

  for (int t = 0; t < LT; ++t) {
    const __bf16* hprev = hbuf + (size_t)(t & 1) * (LB * LH);
    __bf16*       hnext = hbuf + (size_t)((t + 1) & 1) * (LB * LH);

    v8f acc[4] = {};

    // K = 0..255 : x part (fp32 -> bf16 via v_cvt_pk_bf16_f32)
    const float* px = px0 + (size_t)t * LD;
    __builtin_prefetch(px + LD, 0, 1);                   // next timestep's x slice
    #pragma unroll 1
    for (int kk = 0; kk < 8; ++kk) {
      v16bf a;
      #pragma unroll
      for (int e = 0; e < 8; ++e) a[e]     = (__bf16)px[kk * 32 + e];
      #pragma unroll
      for (int e = 0; e < 8; ++e) a[e + 8] = (__bf16)px[kk * 32 + 16 + e];
      const __bf16* lb = &ldsW[(size_t)kk * 4 * 32 * 16];
      // Fetch all four gate B-tiles first so ds loads batch ahead of the WMMAs.
      v16bf btv[4];
      #pragma unroll
      for (int g = 0; g < 4; ++g)
        btv[g] = *(const v16bf*)(lb + ((g * 32 + lane) << 4));
      #pragma unroll
      for (int g = 0; g < 4; ++g)
        acc[g] = __builtin_amdgcn_wmma_f32_16x16x32_bf16(
            false, a, false, btv[g], (short)0, acc[g], false, false);
    }

    // K = 256..767 : h part (skipped at t=0 since h0 == 0)
    if (t > 0) {
      const __bf16* ph = hprev + (size_t)rowA * LH + kaA;
      #pragma unroll 1
      for (int kk = 8; kk < 24; ++kk) {
        const int kh = (kk - 8) * 32;
        v8bf lo = *(const v8bf*)(ph + kh);
        v8bf hi = *(const v8bf*)(ph + kh + 16);
        v16bf a;
        #pragma unroll
        for (int e = 0; e < 8; ++e) { a[e] = lo[e]; a[e + 8] = hi[e]; }
        const __bf16* lb = &ldsW[(size_t)kk * 4 * 32 * 16];
        v16bf btv[4];
        #pragma unroll
        for (int g = 0; g < 4; ++g)
          btv[g] = *(const v16bf*)(lb + ((g * 32 + lane) << 4));
        #pragma unroll
        for (int g = 0; g < 4; ++g)
          acc[g] = __builtin_amdgcn_wmma_f32_16x16x32_bf16(
              false, a, false, btv[g], (short)0, acc[g], false, false);
      }
    }

    // Epilogue: gates -> c,h update. Each lane owns 8 rows x 1 col.
    #pragma unroll
    for (int r = 0; r < 8; ++r) {
      const int row = rowbase + r;
      const float iv = sigm(acc[0][r] + bi);
      const float fv = sigm(acc[1][r] + bff);
      const float gv = ftanh(acc[2][r] + bg);
      const float ov = sigm(acc[3][r] + bo);
      const float cn = fv * creg[r] + iv * gv;
      const float hn = ov * ftanh(cn);
      creg[r] = cn;
      hnext[(size_t)row * LH + col] = (__bf16)hn;
      out[((size_t)t * LB + row) * LH + col] = hn;       // hs[T,B,H] flattened
    }

    // Grid-wide sync: monotonically increasing arrival counter.
    __syncthreads();
    if (tid == 0) {
      __threadfence();                                   // publish h stores
      atomicAdd(cnt, 1u);
      const unsigned target = (unsigned)(NBLK * (t + 1));
      while (__hip_atomic_load(cnt, __ATOMIC_ACQUIRE, __HIP_MEMORY_SCOPE_AGENT) < target) {
        __builtin_amdgcn_s_sleep(1);
      }
    }
    __syncthreads();
    __threadfence();                                     // observe other blocks' h
  }
}

extern "C" void kernel_launch(void* const* d_in, const int* in_sizes, int n_in,
                              void* d_out, int out_size, void* d_ws, size_t ws_size,
                              hipStream_t stream) {
  const float* x    = (const float*)d_in[0];
  const float* W_ih = (const float*)d_in[1];
  const float* W_hh = (const float*)d_in[2];
  const float* b_ih = (const float*)d_in[3];
  const float* b_hh = (const float*)d_in[4];
  float* out = (float*)d_out;

  // Workspace: [0,4) sync counter; [256, 256+262144) h ping-pong (bf16 x2)
  char* ws = (char*)d_ws;
  unsigned* cnt = (unsigned*)ws;
  __bf16*  hbuf = (__bf16*)(ws + 256);

  lstm_init<<<1, 64, 0, stream>>>(cnt);
  lstm_persist<<<NBLK, NTHR, LDS_BYTES, stream>>>(
      x, W_ih, W_hh, b_ih, b_hh, hbuf, out, cnt);

  (void)in_sizes; (void)n_in; (void)out_size; (void)ws_size;
}
